// LiGRUModel_52347061404178
// MI455X (gfx1250) — matrix operand order repaired and checked by reading, not compiled, not run on hardware
//
#include <hip/hip_runtime.h>
#include <hip/hip_bf16.h>
#include <cmath>

// ---------------------------------------------------------------------------
// LiGRU bidirectional 3-layer + FC for MI455X (gfx1250, wave32, WMMA).
// - All GEMMs: v_wmma_f32_16x16x32_f16 (fp16 in, fp32 acc).
// - Input projections: one big WMMA GEMM per layer.
// - Recurrence: ONE persistent kernel per layer; 500 time steps inside with a
//   device-wide generation barrier between steps. U tiles are staged in LDS
//   (32 KB/block of the 320 KB WGP LDS) so the B operand of every recurrent
//   WMMA is fed from ds_load instead of L2.
// ---------------------------------------------------------------------------

typedef __attribute__((ext_vector_type(16))) _Float16 v16h;
typedef __attribute__((ext_vector_type(8)))  _Float16 v8h;
typedef __attribute__((ext_vector_type(8)))  float    v8f;

#define WMMA_F16(a, b, c) \
  __builtin_amdgcn_wmma_f32_16x16x32_f16(false, (a), false, (b), (short)0, (c), false, false)

// A-fragment (16x32, MxK), row-major A [.., ld=K]:
// lane l<16: row m0+l, K = k0..k0+15
// lane l>=16: row m0+l-16, K = k0+16..k0+31
__device__ __forceinline__ v16h load_a_frag(const _Float16* __restrict__ A, int ld,
                                            int m0, int k0, int lane) {
  int row = m0 + (lane & 15);
  int kb  = k0 + ((lane >> 4) << 4);
  return *(const v16h*)(A + (size_t)row * ld + kb);
}

// B-fragment (32x16, KxN), B stored transposed row-major Bt [N, ld=K]:
// lane l<16: col n0+l, K = k0..k0+15 ; lane l>=16: col, K = k0+16..k0+31
__device__ __forceinline__ v16h load_b_frag(const _Float16* __restrict__ Bt, int ld,
                                            int n0, int k0, int lane) {
  int col = n0 + (lane & 15);
  int ks  = k0 + ((lane >> 4) << 4);
  return *(const v16h*)(Bt + (size_t)col * ld + ks);
}

// Same B-fragment pattern but reading a 16x512 tile resident in LDS.
__device__ __forceinline__ v16h load_b_frag_lds(const _Float16* Us, int k0, int lane) {
  int col = lane & 15;
  int ks  = k0 + ((lane >> 4) << 4);
  return *(const v16h*)(Us + col * 512 + ks);
}

// ---------------------------------------------------------------------------
// Generic f16 GEMM: C[M,N] = A[M,K] @ B[K,N], B given transposed [N,K].
// One wave computes a 16x64 tile. M%16==0, N%64==0, K%32==0.
// ---------------------------------------------------------------------------
__global__ void __launch_bounds__(256)
wmma_gemm_f16(const _Float16* __restrict__ A, const _Float16* __restrict__ Bt,
              _Float16* __restrict__ C, int M, int N, int K) {
  int wave = (int)((blockIdx.x * blockDim.x + threadIdx.x) >> 5);
  int lane = (int)(threadIdx.x & 31);
  int ngroups = N >> 6;
  int mt = wave / ngroups;
  int ng = wave % ngroups;
  if ((mt << 4) >= M) return;           // wave-uniform
  int m0 = mt << 4, n0 = ng << 6;
  v8f acc[4] = {};
  for (int k0 = 0; k0 < K; k0 += 32) {
    if (k0 + 32 < K)
      __builtin_prefetch(A + (size_t)(m0 + (lane & 15)) * K + k0 + 32, 0, 3);
    v16h a = load_a_frag(A, K, m0, k0, lane);
#pragma unroll
    for (int j = 0; j < 4; ++j) {
      v16h b = load_b_frag(Bt, K, n0 + (j << 4), k0, lane);
      acc[j] = WMMA_F16(a, b, acc[j]);
    }
  }
  int lo = lane & 15, hi = lane >> 4;
#pragma unroll
  for (int j = 0; j < 4; ++j) {
#pragma unroll
    for (int v = 0; v < 8; ++v) {
      int r = m0 + v + (hi << 3);
      C[(size_t)r * N + n0 + (j << 4) + lo] = (_Float16)acc[j][v];
    }
  }
}

// ---------------------------------------------------------------------------
// Device-wide barrier (generation counter in global memory). 32 co-resident
// blocks; release fence before arrival, acquire fence after release observed.
// ---------------------------------------------------------------------------
__device__ __forceinline__ void grid_barrier(unsigned* count, volatile unsigned* gen,
                                             unsigned nblk) {
  __threadfence();          // make this block's h stores visible device-wide
  __syncthreads();
  if (threadIdx.x == 0) {
    unsigned g = *gen;
    if (atomicAdd(count, 1u) + 1u == nblk) {
      atomicExch(count, 0u);
      __threadfence();
      *gen = g + 1u;        // release
    } else {
      while (*gen == g) __builtin_amdgcn_s_sleep(2);
    }
  }
  __syncthreads();
  __threadfence();          // acquire: discard stale cached h
}

// ---------------------------------------------------------------------------
// Persistent LiGRU scan for one layer. Grid: 32 blocks x 256 threads.
// Block nt owns gate columns [16nt,16nt+16) (both a and z halves); its two
// 16x512 U tiles live in LDS. Wave w in the block owns batch rows [16w,16w+16).
// Each step: gates = wx[:,t,:] + h @ U (WMMA, B from LDS), activation update,
// scatter h' into xb_next (bidirectional concat + time reversal), grid barrier.
// ---------------------------------------------------------------------------
__global__ void __launch_bounds__(256)
ligru_scan(const _Float16* __restrict__ Ut,   // [1024, 512] transposed U
           const _Float16* __restrict__ wx,   // [128*T, 1024]
           _Float16* __restrict__ hA,         // [128,512] (h at even t)
           _Float16* __restrict__ hB,         // [128,512] (h at odd t)
           _Float16* __restrict__ xb_next,    // [128*T, 1024]
           unsigned* __restrict__ bar_count,
           volatile unsigned* bar_gen,
           int T) {
  __shared__ _Float16 Us[2 * 16 * 512];       // 32 KB: a-tile then z-tile
  const int nt   = (int)blockIdx.x;           // 0..31
  const int lane = (int)(threadIdx.x & 31);
  const int m0   = (int)((threadIdx.x >> 5) << 4);   // wave -> batch row tile
  const int n0   = nt << 4;

  // Stage this block's U tiles into LDS (rows n0.. and 512+n0.. of Ut).
  {
    const v8h* srcA = (const v8h*)(Ut + (size_t)n0 * 512);
    const v8h* srcZ = (const v8h*)(Ut + (size_t)(512 + n0) * 512);
    v8h* dstA = (v8h*)Us;
    v8h* dstZ = (v8h*)(Us + 16 * 512);
    for (int i = (int)threadIdx.x; i < (16 * 512) / 8; i += 256) {
      dstA[i] = srcA[i];
      dstZ[i] = srcZ[i];
    }
  }
  __syncthreads();

  for (int t = 0; t < T; ++t) {
    const _Float16* hc = (t & 1) ? hB : hA;
    _Float16*       hn = (t & 1) ? hA : hB;

    v8f accA = {}, accZ = {};
    for (int k0 = 0; k0 < 512; k0 += 32) {
      v16h a  = load_a_frag(hc, 512, m0, k0, lane);
      v16h bA = load_b_frag_lds(Us, k0, lane);
      v16h bZ = load_b_frag_lds(Us + 16 * 512, k0, lane);
      accA = WMMA_F16(a, bA, accA);
      accZ = WMMA_F16(a, bZ, accZ);
    }

    int lo = lane & 15, hi = lane >> 4;
    int col = n0 + lo;
#pragma unroll
    for (int v = 0; v < 8; ++v) {
      int m = m0 + v + (hi << 3);
      size_t wrow = (size_t)m * T + t;
      float ga = accA[v] + (float)wx[wrow * 1024 + col];
      float gz = accZ[v] + (float)wx[wrow * 1024 + col + 512];
      float z    = 1.0f / (1.0f + __expf(-gz));
      float hold = (float)hc[(size_t)m * 512 + col];
      float hv   = z * hold + (1.0f - z) * tanhf(ga);
      _Float16 hh = (_Float16)hv;
      hn[(size_t)m * 512 + col] = hh;
      // Bidirectional output assembly into next layer's input:
      //  fwd rows (m<64): y[b, t, col]; bwd rows: y[b, T-1-t, 512+col];
      //  plus time-reversed replica at batch rows 64..127.
      int b  = (m < 64) ? m : (m - 64);
      int tf = (m < 64) ? t : (T - 1 - t);
      int f  = (m < 64) ? col : (col + 512);
      xb_next[((size_t)b * T + tf) * 1024 + f] = hh;
      xb_next[((size_t)(64 + b) * T + (T - 1 - tf)) * 1024 + f] = hh;
    }

    if (t + 1 < T) grid_barrier(bar_count, bar_gen, (unsigned)gridDim.x);
  }
}

// ---------------------------------------------------------------------------
// Final FC: out[32000,40] f32 = y[32000,1024] @ Wfc[1024,40] + bfc.
// ---------------------------------------------------------------------------
__global__ void __launch_bounds__(256)
fc_wmma(const _Float16* __restrict__ A, const _Float16* __restrict__ Wt,
        const float* __restrict__ bias, float* __restrict__ out, int M) {
  int wave = (int)((blockIdx.x * blockDim.x + threadIdx.x) >> 5);
  int lane = (int)(threadIdx.x & 31);
  int mt = wave / 3, ntile = wave % 3;
  if ((mt << 4) >= M) return;          // wave-uniform
  int m0 = mt << 4, n0 = ntile << 4;
  v8f acc = {};
  for (int k0 = 0; k0 < 1024; k0 += 32) {
    v16h a = load_a_frag(A, 1024, m0, k0, lane);
    v16h b = load_b_frag(Wt, 1024, n0, k0, lane);
    acc = WMMA_F16(a, b, acc);
  }
  int lo = lane & 15, hi = lane >> 4;
  int col = n0 + lo;
  if (col < 40) {
#pragma unroll
    for (int v = 0; v < 8; ++v) {
      int r = m0 + v + (hi << 3);
      out[(size_t)r * 40 + col] = acc[v] + bias[col];
    }
  }
}

// ---------------------------------------------------------------------------
// Conversion / setup kernels
// ---------------------------------------------------------------------------
__global__ void convert_transpose(const float* __restrict__ src, _Float16* __restrict__ dst,
                                  int K, int N, int Kpad, int Npad) {
  int idx = (int)(blockIdx.x * blockDim.x + threadIdx.x);
  int total = Kpad * Npad;
  if (idx >= total) return;
  int n = idx / Kpad, k = idx - n * Kpad;
  float v = (k < K && n < N) ? src[(size_t)k * N + n] : 0.0f;
  dst[idx] = (_Float16)v;
}

__global__ void build_xb0(const float* __restrict__ x, _Float16* __restrict__ xb, int T) {
  size_t idx = (size_t)blockIdx.x * blockDim.x + threadIdx.x;
  size_t total = (size_t)128 * T * 96;
  if (idx >= total) return;
  int k  = (int)(idx % 96);
  int t  = (int)((idx / 96) % T);
  int bp = (int)(idx / ((size_t)96 * T));
  int b  = (bp < 64) ? bp : bp - 64;
  int tt = (bp < 64) ? t : (T - 1 - t);
  float v = (k < 89) ? x[((size_t)b * T + tt) * 89 + k] : 0.0f;
  xb[idx] = (_Float16)v;
}

__global__ void zero_half(_Float16* __restrict__ p, int n) {
  int i = (int)(blockIdx.x * blockDim.x + threadIdx.x);
  if (i < n) p[i] = (_Float16)0.0f;
}

__global__ void init_barrier(unsigned* __restrict__ p) {
  if (threadIdx.x < 2) p[threadIdx.x] = 0u;
}

// ---------------------------------------------------------------------------
extern "C" void kernel_launch(void* const* d_in, const int* in_sizes, int n_in,
                              void* d_out, int out_size, void* d_ws, size_t ws_size,
                              hipStream_t stream) {
  (void)in_sizes; (void)n_in; (void)out_size; (void)ws_size;
  const int T = 500;

  const float* x   = (const float*)d_in[0];
  // d_in[1] = lengths (all ones, unused)
  const float* W0  = (const float*)d_in[2];
  const float* U0  = (const float*)d_in[3];
  const float* W1  = (const float*)d_in[4];
  const float* U1  = (const float*)d_in[5];
  const float* W2  = (const float*)d_in[6];
  const float* U2  = (const float*)d_in[7];
  const float* Wfc = (const float*)d_in[8];
  const float* bfc = (const float*)d_in[9];
  float* out = (float*)d_out;

  // ---- workspace layout (f16, 32B-aligned chunks) ----
  _Float16* ws = (_Float16*)d_ws;
  size_t off = 0;
  auto alloc = [&](size_t n) {
    _Float16* p = ws + off;
    off += (n + 15) & ~(size_t)15;
    return p;
  };
  _Float16* xbA = alloc((size_t)128 * T * 1024);
  _Float16* xbB = alloc((size_t)128 * T * 1024);
  _Float16* wx  = alloc((size_t)128 * T * 1024);
  _Float16* xb0 = alloc((size_t)128 * T * 96);
  _Float16* hA  = alloc((size_t)128 * 512);
  _Float16* hB  = alloc((size_t)128 * 512);
  _Float16* W0t = alloc((size_t)1024 * 96);
  _Float16* W1t = alloc((size_t)1024 * 1024);
  _Float16* W2t = alloc((size_t)1024 * 1024);
  _Float16* U0t = alloc((size_t)1024 * 512);
  _Float16* U1t = alloc((size_t)1024 * 512);
  _Float16* U2t = alloc((size_t)1024 * 512);
  _Float16* Wft = alloc((size_t)48 * 1024);
  unsigned* bar = (unsigned*)alloc(16);       // [0]=count, [1]=generation

  // ---- weight conversion (f32 -> f16 transposed, padded) ----
  auto cvt = [&](const float* s, _Float16* d, int K, int N, int Kp, int Np) {
    int tot = Kp * Np;
    convert_transpose<<<(tot + 255) / 256, 256, 0, stream>>>(s, d, K, N, Kp, Np);
  };
  cvt(W0, W0t,   89, 1024,   96, 1024);
  cvt(W1, W1t, 1024, 1024, 1024, 1024);
  cvt(W2, W2t, 1024, 1024, 1024, 1024);
  cvt(U0, U0t,  512, 1024,  512, 1024);
  cvt(U1, U1t,  512, 1024,  512, 1024);
  cvt(U2, U2t,  512, 1024,  512, 1024);
  cvt(Wfc, Wft, 1024, 40, 1024, 48);
  {
    size_t tot = (size_t)128 * T * 96;
    build_xb0<<<(unsigned)((tot + 255) / 256), 256, 0, stream>>>(x, xb0, T);
  }

  const int M = 128 * T;  // 64000 GEMM rows per layer
  struct Layer { const _Float16* xin; int K; const _Float16* Wt; const _Float16* Ut; _Float16* xout; };
  Layer Ls[3] = {
    { xb0,   96, W0t, U0t, xbA },
    { xbA, 1024, W1t, U1t, xbB },
    { xbB, 1024, W2t, U2t, xbA },
  };

  for (int L = 0; L < 3; ++L) {
    // wx = xb @ W   ([64000,K] x [K,1024])
    int waves  = (M / 16) * (1024 / 64);
    int blocks = waves / 8;
    wmma_gemm_f16<<<blocks, 256, 0, stream>>>(Ls[L].xin, Ls[L].Wt, wx, M, 1024, Ls[L].K);
    // h0 = 0; reset device-wide barrier state
    zero_half<<<(128 * 512 + 255) / 256, 256, 0, stream>>>(hA, 128 * 512);
    init_barrier<<<1, 32, 0, stream>>>(bar);
    // whole 500-step scan in one persistent launch
    ligru_scan<<<32, 256, 0, stream>>>(Ls[L].Ut, wx, hA, hB, Ls[L].xout,
                                       bar, (volatile unsigned*)(bar + 1), T);
  }

  // out = y @ Wfc + bfc  (y = rows 0..31999 of xbA, i.e. [64,T,1024])
  {
    int waves  = (32000 / 16) * 3;
    int blocks = (waves * 32 + 255) / 256;
    fc_wmma<<<blocks, 256, 0, stream>>>(xbA, Wft, bfc, out, 32000);
  }
}
